// Decoder_33019708571928
// MI455X (gfx1250) — matrix operand order repaired and checked by reading, not compiled, not run on hardware
//
#include <hip/hip_runtime.h>
#include <math.h>

typedef __bf16 v16bf __attribute__((ext_vector_type(16)));
typedef __bf16 v8bf  __attribute__((ext_vector_type(8)));
typedef __bf16 v4bf  __attribute__((ext_vector_type(4)));
typedef float  v8f   __attribute__((ext_vector_type(8)));
typedef float  v4f   __attribute__((ext_vector_type(4)));

#define B_SZ   128
#define S_SZ   1024
#define U_SZ   1024
#define IN_DIM 64
#define MT     64          // rows of (B*S) per block in the score kernel
#define LDK    1032        // padded LDS row stride (bf16 elems): 516 dwords -> bank shift 4/row

__device__ __forceinline__ float sigmoidf_(float x) { return 1.0f / (1.0f + __expf(-x)); }

// Branch-free fast tanh: native v_tanh_f32 if the builtin exists, else
// 1 - 2/(exp(2x)+1)  (exp->inf => 1, exp->0 => -1; no NaN, no EXEC branches).
__device__ __forceinline__ float tanh_fast(float x) {
#if __has_builtin(__builtin_amdgcn_tanh_f32)
    return __builtin_amdgcn_tanh_f32(x);
#elif __has_builtin(__builtin_amdgcn_tanhf)
    return __builtin_amdgcn_tanhf(x);
#else
    float e = __expf(2.0f * x);
    return 1.0f - 2.0f / (e + 1.0f);
#endif
}

// ---------------------------------------------------------------------------
// Prep: W1 (K x N, f32) -> W1t (N x K, bf16)   (coalesced 2B writes)
// ---------------------------------------------------------------------------
__global__ void k_transpose_w1(const float* __restrict__ W1, __bf16* __restrict__ W1t) {
    int g = blockIdx.x * blockDim.x + threadIdx.x;   // 1M threads
    int n = g >> 10;
    int k = g & 1023;
    W1t[n * 1024 + k] = (__bf16)W1[k * 1024 + n];
}

// ---------------------------------------------------------------------------
// Prep: hb[b,n] = b1[n] + b2[n] + hidden[b,:] @ W2[:,n]
// ---------------------------------------------------------------------------
__global__ void k_hb(const float* __restrict__ hidden, const float* __restrict__ W2,
                     const float* __restrict__ b1, const float* __restrict__ b2,
                     float* __restrict__ hb) {
    __shared__ float hrow[U_SZ];
    int b = blockIdx.x;
    int tid = threadIdx.x;
    for (int k = tid; k < U_SZ; k += 256) hrow[k] = hidden[b * U_SZ + k];
    __syncthreads();
    int n = blockIdx.y * 256 + tid;
    float acc = b1[n] + b2[n];
    for (int k = 0; k < U_SZ; ++k) acc += hrow[k] * W2[k * U_SZ + n];
    hb[b * U_SZ + n] = acc;
}

// ---------------------------------------------------------------------------
// Prep: hzr[b,j] = hidden[b,:] @ Wr[:, j]   for j in [0, 2U)   (z and r gates)
// ---------------------------------------------------------------------------
__global__ void k_hzr(const float* __restrict__ hidden, const float* __restrict__ Wr,
                      float* __restrict__ hzr) {
    __shared__ float hrow[U_SZ];
    int b = blockIdx.x;
    int tid = threadIdx.x;
    for (int k = tid; k < U_SZ; k += 256) hrow[k] = hidden[b * U_SZ + k];
    __syncthreads();
    int j = blockIdx.y * 256 + tid;            // 0 .. 2047
    float acc = 0.0f;
    for (int k = 0; k < U_SZ; ++k) acc += hrow[k] * Wr[k * (3 * U_SZ) + j];
    hzr[b * (2 * U_SZ) + j] = acc;
}

// ---------------------------------------------------------------------------
// MAIN: fused  score[b,s] = V . tanh(enc[b,s,:]@W1 + hb[b,:]) + bV
//   block = 256 threads (8 waves), covers MT=64 consecutive (b,s) rows.
//   A tile (64 x 1024 f32, contiguous) staged to LDS as bf16 once.
//   Wave owns col-tiles ct = wave, wave+8, ...; per B frag (loaded once from
//   global) it runs 4 WMMAs against the 4 row tiles. All 4 A frags are
//   materialized in distinct registers BEFORE the WMMAs so the 8 ds_load_b128
//   issue as a clause with staggered s_wait_dscnt instead of full drains.
// ---------------------------------------------------------------------------
__global__ void k_score(const float* __restrict__ enc, const __bf16* __restrict__ W1t,
                        const float* __restrict__ hb, const float* __restrict__ Vv,
                        const float* __restrict__ bV, float* __restrict__ scoreg) {
    __shared__ __bf16 Alds[MT * LDK];     // 132096 B
    __shared__ float  score_s[MT];

    const int tid  = threadIdx.x;
    const int wave = tid >> 5;
    const int lane = tid & 31;
    const int l16  = lane & 15;
    const bool hi  = lane >= 16;

    // ---- stage A tile: 64x1024 f32 -> bf16 LDS (vectorized, padded rows) ----
    const size_t base = (size_t)blockIdx.x * MT * 1024;
    for (int i = tid * 4; i < MT * 1024; i += 256 * 4) {
        int row = i >> 10, col = i & 1023;
        v4f v = *reinterpret_cast<const v4f*>(enc + base + i);
        v4bf o;
        o[0] = (__bf16)v[0]; o[1] = (__bf16)v[1]; o[2] = (__bf16)v[2]; o[3] = (__bf16)v[3];
        *reinterpret_cast<v4bf*>(&Alds[row * LDK + col]) = o;
    }
    if (tid < MT) score_s[tid] = bV[0];
    __syncthreads();

    const int b = blockIdx.x >> 4;                 // 16 blocks per batch (1024/64)
    const float* hbrow = hb + b * U_SZ;

    const __bf16* Arow[4];
#pragma unroll
    for (int rt = 0; rt < 4; ++rt)
        Arow[rt] = &Alds[(rt * 16 + l16) * LDK + (hi ? 16 : 0)];

    float vp[4][8];
#pragma unroll
    for (int rt = 0; rt < 4; ++rt)
#pragma unroll
        for (int i = 0; i < 8; ++i) vp[rt][i] = 0.0f;

    const v8f czero = {};

    for (int ct = wave; ct < 64; ct += 8) {
        const int n = ct * 16 + l16;
        const __bf16* Bbase = W1t + (size_t)n * 1024 + (hi ? 16 : 0);
        const float hbn = hbrow[n];     // issue early: overlaps the K loop
        const float Vn  = Vv[n];

        v8f c[4];
        c[0] = czero; c[1] = czero; c[2] = czero; c[3] = czero;

#pragma unroll 2
        for (int kk = 0; kk < 32; ++kk) {
            const int kbase = kk * 32;
            v16bf bb = *reinterpret_cast<const v16bf*>(Bbase + kbase);

            // batch all A-frag LDS loads first (distinct registers)
            v8bf a0[4], a1[4];
#pragma unroll
            for (int rt = 0; rt < 4; ++rt) {
                a0[rt] = *reinterpret_cast<const v8bf*>(Arow[rt] + kbase);
                a1[rt] = *reinterpret_cast<const v8bf*>(Arow[rt] + kbase + 8);
            }
#pragma unroll
            for (int rt = 0; rt < 4; ++rt) {
                v16bf a = __builtin_shufflevector(a0[rt], a1[rt],
                                                  0,1,2,3,4,5,6,7,8,9,10,11,12,13,14,15);
                c[rt] = __builtin_amdgcn_wmma_f32_16x16x32_bf16(false, a, false, bb,
                                                                (short)0, c[rt],
                                                                false, false);
            }
        }

#pragma unroll
        for (int rt = 0; rt < 4; ++rt)
#pragma unroll
            for (int i = 0; i < 8; ++i)
                vp[rt][i] += tanh_fast(c[rt][i] + hbn) * Vn;
    }

    // reduce over the 16 N-lanes (xor masks 1..8 stay within each half-wave)
#pragma unroll
    for (int off = 8; off >= 1; off >>= 1)
#pragma unroll
        for (int rt = 0; rt < 4; ++rt)
#pragma unroll
            for (int i = 0; i < 8; ++i)
                vp[rt][i] += __shfl_xor(vp[rt][i], off, 32);

    if (l16 == 0) {
#pragma unroll
        for (int rt = 0; rt < 4; ++rt) {
            const int mbase = rt * 16 + (hi ? 8 : 0);
#pragma unroll
            for (int i = 0; i < 8; ++i) atomicAdd(&score_s[mbase + i], vp[rt][i]);
        }
    }
    __syncthreads();
    if (tid < MT) scoreg[(size_t)blockIdx.x * MT + tid] = score_s[tid];
}

// ---------------------------------------------------------------------------
// Softmax over S + context[b,e] = sum_s w[s]*enc[b,s,e]; writes attn weights.
// grid (B, 4), block 256; each block covers 256 e-columns.
// ---------------------------------------------------------------------------
__global__ void k_softmax_ctx(const float* __restrict__ scoreg, const float* __restrict__ enc,
                              float* __restrict__ ctx, float* __restrict__ attn_out) {
    __shared__ float w[S_SZ];
    __shared__ float red[256];
    __shared__ float bcast;
    int b = blockIdx.x, tid = threadIdx.x;

    for (int s = tid; s < S_SZ; s += 256) w[s] = scoreg[b * S_SZ + s];
    __syncthreads();

    float m = -3.4e38f;
    for (int s = tid; s < S_SZ; s += 256) m = fmaxf(m, w[s]);
    red[tid] = m; __syncthreads();
    for (int off = 128; off > 0; off >>= 1) {
        if (tid < off) red[tid] = fmaxf(red[tid], red[tid + off]);
        __syncthreads();
    }
    if (tid == 0) bcast = red[0];
    __syncthreads();
    float M = bcast;
    __syncthreads();

    float sum = 0.0f;
    for (int s = tid; s < S_SZ; s += 256) { float e = __expf(w[s] - M); w[s] = e; sum += e; }
    red[tid] = sum; __syncthreads();
    for (int off = 128; off > 0; off >>= 1) {
        if (tid < off) red[tid] += red[tid + off];
        __syncthreads();
    }
    if (tid == 0) bcast = 1.0f / red[0];
    __syncthreads();
    float inv = bcast;
    for (int s = tid; s < S_SZ; s += 256) w[s] *= inv;
    __syncthreads();

    if (blockIdx.y == 0)
        for (int s = tid; s < S_SZ; s += 256) attn_out[b * S_SZ + s] = w[s];

    int e = blockIdx.y * 256 + tid;
    float acc = 0.0f;
    const float* ebase = enc + (size_t)b * S_SZ * U_SZ + e;
    for (int s = 0; s < S_SZ; ++s) acc += w[s] * ebase[(size_t)s * U_SZ];
    ctx[b * U_SZ + e] = acc;
}

// ---------------------------------------------------------------------------
// xg[b,j] = bg[j] + [ctx, x][b,:] @ Wk[:,j]   (K=1088, N=3072)
// ---------------------------------------------------------------------------
__global__ void k_xg(const float* __restrict__ ctx, const float* __restrict__ x,
                     const float* __restrict__ Wk, const float* __restrict__ bg,
                     float* __restrict__ xg) {
    __shared__ float gin[U_SZ + IN_DIM];
    int b = blockIdx.x, tid = threadIdx.x;
    for (int k = tid; k < U_SZ; k += 256) gin[k] = ctx[b * U_SZ + k];
    if (tid < IN_DIM) gin[U_SZ + tid] = x[b * IN_DIM + tid];
    __syncthreads();
    int j = blockIdx.y * 256 + tid;            // 0 .. 3071
    float acc = bg[j];
    for (int k = 0; k < U_SZ + IN_DIM; ++k) acc += gin[k] * Wk[k * (3 * U_SZ) + j];
    xg[b * (3 * U_SZ) + j] = acc;
}

// ---------------------------------------------------------------------------
// gates: z = sig(xg_z + hz); r = sig(xg_r + hr); rh = r * hidden
// ---------------------------------------------------------------------------
__global__ void k_gates(const float* __restrict__ xg, const float* __restrict__ hzr,
                        const float* __restrict__ hidden,
                        float* __restrict__ zbuf, float* __restrict__ rhbuf) {
    int b = blockIdx.x;
    int u = blockIdx.y * 256 + threadIdx.x;
    float z = sigmoidf_(xg[b * (3 * U_SZ) + u]            + hzr[b * (2 * U_SZ) + u]);
    float r = sigmoidf_(xg[b * (3 * U_SZ) + U_SZ + u]     + hzr[b * (2 * U_SZ) + U_SZ + u]);
    zbuf[b * U_SZ + u]  = z;
    rhbuf[b * U_SZ + u] = r * hidden[b * U_SZ + u];
}

// ---------------------------------------------------------------------------
// hh = tanh(xg_h + rh @ Wr[:,2U:]); state = z*h + (1-z)*hh  -> d_out state
// ---------------------------------------------------------------------------
__global__ void k_state(const float* __restrict__ xg, const float* __restrict__ rhbuf,
                        const float* __restrict__ Wr, const float* __restrict__ zbuf,
                        const float* __restrict__ hidden, float* __restrict__ state_out) {
    __shared__ float rh[U_SZ];
    int b = blockIdx.x, tid = threadIdx.x;
    for (int k = tid; k < U_SZ; k += 256) rh[k] = rhbuf[b * U_SZ + k];
    __syncthreads();
    int u = blockIdx.y * 256 + tid;
    float acc = xg[b * (3 * U_SZ) + 2 * U_SZ + u];
    for (int k = 0; k < U_SZ; ++k) acc += rh[k] * Wr[k * (3 * U_SZ) + 2 * U_SZ + u];
    float hh = tanh_fast(acc);
    float z  = zbuf[b * U_SZ + u];
    state_out[b * U_SZ + u] = z * hidden[b * U_SZ + u] + (1.0f - z) * hh;
}

// ---------------------------------------------------------------------------
// out[b,j] = bfc[j] + state[b,:] @ Wfc[:,j]
// ---------------------------------------------------------------------------
__global__ void k_fc(const float* __restrict__ state, const float* __restrict__ Wfc,
                     const float* __restrict__ bfc, float* __restrict__ out) {
    __shared__ float st[U_SZ];
    int b = blockIdx.x, tid = threadIdx.x;   // 64 threads
    for (int k = tid; k < U_SZ; k += 64) st[k] = state[b * U_SZ + k];
    __syncthreads();
    float acc = bfc[tid];
    for (int k = 0; k < U_SZ; ++k) acc += st[k] * Wfc[k * IN_DIM + tid];
    out[b * IN_DIM + tid] = acc;
}

// ---------------------------------------------------------------------------
extern "C" void kernel_launch(void* const* d_in, const int* in_sizes, int n_in,
                              void* d_out, int out_size, void* d_ws, size_t ws_size,
                              hipStream_t stream) {
    const float* x      = (const float*)d_in[0];
    const float* hidden = (const float*)d_in[1];
    const float* enc    = (const float*)d_in[2];
    const float* W1     = (const float*)d_in[3];
    const float* b1     = (const float*)d_in[4];
    const float* W2     = (const float*)d_in[5];
    const float* b2     = (const float*)d_in[6];
    const float* Vv     = (const float*)d_in[7];
    const float* bV     = (const float*)d_in[8];
    const float* Wk     = (const float*)d_in[9];
    const float* Wr     = (const float*)d_in[10];
    const float* bg     = (const float*)d_in[11];
    const float* Wfc    = (const float*)d_in[12];
    const float* bfc    = (const float*)d_in[13];

    float* out_fc    = (float*)d_out;                       // 128*64
    float* out_state = out_fc + B_SZ * IN_DIM;              // 128*1024
    float* out_attn  = out_state + B_SZ * U_SZ;             // 128*1024

    char* ws = (char*)d_ws;
    __bf16* w1t  = (__bf16*)(ws + 0);                       // 2 MB
    float* score = (float*)(ws + (2u << 20));               // 512 KB
    float* hb    = (float*)(ws + (2u << 20) + (512u << 10));
    float* hzr   = (float*)(ws + (3u << 20));               // 1 MB
    float* ctx   = (float*)(ws + (4u << 20));               // 512 KB
    float* xg    = (float*)(ws + (4u << 20) + (512u << 10));// 1.5 MB
    float* zbuf  = (float*)(ws + (6u << 20));               // 512 KB
    float* rhbuf = (float*)(ws + (6u << 20) + (512u << 10));// 512 KB

    // prep (independent of attention result)
    k_transpose_w1<<<(1024 * 1024) / 256, 256, 0, stream>>>(W1, w1t);
    k_hb<<<dim3(B_SZ, 4), 256, 0, stream>>>(hidden, W2, b1, b2, hb);
    k_hzr<<<dim3(B_SZ, 8), 256, 0, stream>>>(hidden, Wr, hzr);

    // main fused attention-score GEMM (bf16 WMMA)
    k_score<<<(B_SZ * S_SZ) / MT, 256, 0, stream>>>(enc, w1t, hb, Vv, bV, score);

    // softmax + context (+ attention weights output)
    k_softmax_ctx<<<dim3(B_SZ, 4), 256, 0, stream>>>(score, enc, ctx, out_attn);

    // GRU step
    k_xg<<<dim3(B_SZ, 12), 256, 0, stream>>>(ctx, x, Wk, bg, xg);
    k_gates<<<dim3(B_SZ, 4), 256, 0, stream>>>(xg, hzr, hidden, zbuf, rhbuf);
    k_state<<<dim3(B_SZ, 4), 256, 0, stream>>>(xg, rhbuf, Wr, zbuf, hidden, out_state);

    // output projection
    k_fc<<<B_SZ, 64, 0, stream>>>(out_state, Wfc, bfc, out_fc);
}
